// AutoRegressiveGraphConvLayer_43018392437144
// MI455X (gfx1250) — matrix operand order repaired and checked by reading, not compiled, not run on hardware
//
#include <hip/hip_runtime.h>
#include <hip/hip_bf16.h>

typedef _Float16 h8   __attribute__((ext_vector_type(8)));
typedef _Float16 h16  __attribute__((ext_vector_type(16)));
typedef float    f8   __attribute__((ext_vector_type(8)));

#define NN     1024
#define MM     32
#define BB     4
#define NEDGE  32240   // sum_{i=1}^{1023} min(i,32) = 496 + 992*32

__device__ __forceinline__ h16 cat16(h8 lo, h8 hi) {
  return __builtin_shufflevector(lo, hi, 0,1,2,3,4,5,6,7,8,9,10,11,12,13,14,15);
}

// ---------------------------------------------------------------------------
// Setup kernels: f32->f16 conversion, weight transpose+convert, edge table
// ---------------------------------------------------------------------------
__global__ void cvt_f16(const float* __restrict__ x, _Float16* __restrict__ y, long n) {
  long i = (long)blockIdx.x * 256 + threadIdx.x;
  if (i < n) y[i] = (_Float16)x[i];
}

// W[fin][fout] (f32) -> Wt[fout][fin] (f16)
__global__ void transpose_w(const float* __restrict__ W, _Float16* __restrict__ Wt,
                            int fin, int fout) {
  int idx = blockIdx.x * 256 + threadIdx.x;
  if (idx >= fin * fout) return;
  int n = idx / fin, k = idx - n * fin;
  Wt[idx] = (_Float16)W[(size_t)k * fout + n];
}

// per-edge (src,dst) node indices
__global__ void build_sd(int2* __restrict__ sd) {
  int e = blockIdx.x * 256 + threadIdx.x;
  if (e >= NEDGE) return;
  int i, j;
  if (e < 496) {                       // triangular region: K(i) = i(i-1)/2
    i = (int)((sqrtf(8.0f * e + 1.0f) + 1.0f) * 0.5f);
    while (i * (i - 1) / 2 > e) --i;
    while (i * (i + 1) / 2 <= e) ++i;
    j = e - i * (i - 1) / 2;
    sd[e] = make_int2(j, i);           // src = j (j0 == 0), dst = i
  } else {                             // dense region: 32 edges per node
    int r = e - 496;
    i = 32 + r / 32;
    j = r % 32;
    sd[e] = make_int2(i - 32 + j, i);
  }
}

// ---------------------------------------------------------------------------
// Fused 2-layer MLP (WMMA f16 -> f32 acc), one 16-row tile per wave.
// MODE 0: node-agg  X=[nodes[src]|edges[e]|nodes[dst]]  (160->128->64, f16 out)
// MODE 1: edge-agg  X=[nodes[src]|edges[e]]             ( 96-> 64->32, f16 out)
// MODE 2: node-out  X=[prevN[r]|nodes[r]]               (128->128->64, f32 out)
// MODE 3: edge-out  X=[prevE[r]|edges[r]]               ( 64-> 64->32, f32 out)
// ---------------------------------------------------------------------------
template<int MODE, int KD, int N0, int N1, bool OUTF32>
__global__ __launch_bounds__(128) void mlp2_kernel(
    const _Float16* __restrict__ nodesF, const _Float16* __restrict__ edgesF,
    const _Float16* __restrict__ prevF,  const int2* __restrict__ sd,
    const _Float16* __restrict__ W0t, const float* __restrict__ b0,
    const _Float16* __restrict__ W1t, const float* __restrict__ b1,
    void* __restrict__ outp)
{
  constexpr int NT0 = N0 / 16, NT1 = N1 / 16, C = KD / 8;
  __shared__ __align__(16) _Float16 sX[4][16 * KD];
  __shared__ __align__(16) _Float16 sY[4][16 * N0];

  const int lane = threadIdx.x & 31;
  const int wave = threadIdx.x >> 5;
  const int tile = blockIdx.x * 4 + wave;
  const long rowbase = (long)tile * 16;
  const int row = lane & 15;      // A-matrix row / B,D column (low 4 bits)
  const int g   = lane >> 4;      // lane half-group

  __builtin_prefetch(W0t, 0, 1);
  __builtin_prefetch(W1t, 0, 1);

  // ---------------- stage gathered X tile into LDS (row major 16 x KD) ------
  {
    long r = rowbase + row;
    const _Float16 *seg0, *seg1, *seg2;
    int S0, S1;
    if constexpr (MODE == 0) {
      int b = (int)(r / NEDGE); int e = (int)(r - (long)b * NEDGE);
      int2 s = sd[e];
      seg0 = nodesF + ((size_t)b * NN + s.x) * 64;
      seg1 = edgesF + ((size_t)b * NEDGE + e) * 32;
      seg2 = nodesF + ((size_t)b * NN + s.y) * 64;
      S0 = 64; S1 = 96;
    } else if constexpr (MODE == 1) {
      int b = (int)(r / NEDGE); int e = (int)(r - (long)b * NEDGE);
      int2 s = sd[e];
      seg0 = nodesF + ((size_t)b * NN + s.x) * 64;
      seg1 = edgesF + ((size_t)b * NEDGE + e) * 32;
      seg2 = seg0; S0 = 64; S1 = KD;
    } else if constexpr (MODE == 2) {
      seg0 = prevF  + (size_t)r * 64;
      seg1 = nodesF + (size_t)r * 64;
      seg2 = seg0; S0 = 64; S1 = KD;
    } else {
      seg0 = prevF  + (size_t)r * 32;
      seg1 = edgesF + (size_t)r * 32;
      seg2 = seg0; S0 = 32; S1 = KD;
    }
#pragma unroll
    for (int c = 0; c < C / 2; ++c) {
      int col = (g * (C / 2) + c) * 8;
      const _Float16* p = (col < S0) ? (seg0 + col)
                        : (col < S1) ? (seg1 + (col - S0))
                                     : (seg2 + (col - S1));
      *(h8*)&sX[wave][row * KD + col] = *(const h8*)p;
    }
  }
  __syncthreads();

  // ---------------- layer 0: X[16xKD] @ W0t^T -> relu -> sY -----------------
  f8 acc[NT0] = {};
#pragma unroll
  for (int kt = 0; kt < KD; kt += 32) {
    // A frag per ISA 16-bit A layout: halves = [K+g*8 .. +8, K+16+g*8 .. +8]
    h16 a = cat16(*(const h8*)&sX[wave][row * KD + kt + g * 8],
                  *(const h8*)&sX[wave][row * KD + kt + 16 + g * 8]);
#pragma unroll
    for (int nt = 0; nt < NT0; ++nt) {
      const _Float16* bp = W0t + ((size_t)(nt * 16 + row)) * KD + kt + g * 8;
      h16 bm = cat16(*(const h8*)bp, *(const h8*)(bp + 16));
      acc[nt] = __builtin_amdgcn_wmma_f32_16x16x32_f16(
          false, a, false, bm, (short)0, acc[nt], false, false);
    }
  }
#pragma unroll
  for (int nt = 0; nt < NT0; ++nt) {
    float bias = b0[nt * 16 + row];
#pragma unroll
    for (int rr = 0; rr < 8; ++rr) {   // D layout: (m = rr + 8*g, n = nt*16 + row)
      float v = acc[nt][rr] + bias;
      v = fmaxf(v, 0.0f);
      sY[wave][(rr + 8 * g) * N0 + nt * 16 + row] = (_Float16)v;
    }
  }
  __syncthreads();

  // ---------------- layer 1: sY[16xN0] @ W1t^T -> relu -> out ---------------
  f8 acc2[NT1] = {};
#pragma unroll
  for (int kt = 0; kt < N0; kt += 32) {
    h16 a = cat16(*(const h8*)&sY[wave][row * N0 + kt + g * 8],
                  *(const h8*)&sY[wave][row * N0 + kt + 16 + g * 8]);
#pragma unroll
    for (int nt = 0; nt < NT1; ++nt) {
      const _Float16* bp = W1t + ((size_t)(nt * 16 + row)) * N0 + kt + g * 8;
      h16 bm = cat16(*(const h8*)bp, *(const h8*)(bp + 16));
      acc2[nt] = __builtin_amdgcn_wmma_f32_16x16x32_f16(
          false, a, false, bm, (short)0, acc2[nt], false, false);
    }
  }
#pragma unroll
  for (int nt = 0; nt < NT1; ++nt) {
    float bias = b1[nt * 16 + row];
#pragma unroll
    for (int rr = 0; rr < 8; ++rr) {
      float v = acc2[nt][rr] + bias;
      v = fmaxf(v, 0.0f);
      size_t o = (size_t)(rowbase + rr + 8 * g) * N1 + nt * 16 + row;
      if constexpr (OUTF32) ((float*)outp)[o] = v;
      else                  ((_Float16*)outp)[o] = (_Float16)v;
    }
  }
}

// ---------------------------------------------------------------------------
// Windowed aggregation: prev_n = mean over node i's edge window;
// prev_e = exclusive prefix mean within the window. Windows are contiguous.
// Block = 96 threads: [0,64) node features, [64,96) edge features.
// ---------------------------------------------------------------------------
__global__ void aggregate_kernel(const _Float16* __restrict__ aggN,
                                 const _Float16* __restrict__ aggE,
                                 _Float16* __restrict__ prevN,
                                 _Float16* __restrict__ prevE)
{
  const int i = blockIdx.x;
  const int b = blockIdx.y;
  const int t = (i < MM) ? i : MM;                 // # incoming edges
  const long base = (i <= MM) ? (long)i * (i - 1) / 2
                              : 496 + (long)(i - MM) * MM;
  const int tid = threadIdx.x;
  if (tid < 64) {
    float s = 0.0f;
    for (int j = 0; j < t; ++j)
      s += (float)aggN[((size_t)b * NEDGE + base + j) * 64 + tid];
    prevN[((size_t)b * NN + i) * 64 + tid] = (_Float16)(t > 0 ? s / (float)t : 0.0f);
  } else {
    const int f = tid - 64;
    float run = 0.0f;
    for (int j = 0; j < t; ++j) {
      size_t idx = ((size_t)b * NEDGE + base + j) * 32 + f;
      prevE[idx] = (_Float16)(j > 0 ? run / (float)j : 0.0f);
      run += (float)aggE[idx];
    }
  }
}

// ---------------------------------------------------------------------------
extern "C" void kernel_launch(void* const* d_in, const int* in_sizes, int n_in,
                              void* d_out, int out_size, void* d_ws, size_t ws_size,
                              hipStream_t stream) {
  (void)in_sizes; (void)n_in; (void)out_size; (void)ws_size;
  const float* in_nodes = (const float*)d_in[0];
  const float* in_edges = (const float*)d_in[1];
  const float* Wan0 = (const float*)d_in[2];  const float* ban0 = (const float*)d_in[3];
  const float* Wan1 = (const float*)d_in[4];  const float* ban1 = (const float*)d_in[5];
  const float* Wln0 = (const float*)d_in[6];  const float* bln0 = (const float*)d_in[7];
  const float* Wln1 = (const float*)d_in[8];  const float* bln1 = (const float*)d_in[9];
  const float* Wae0 = (const float*)d_in[10]; const float* bae0 = (const float*)d_in[11];
  const float* Wae1 = (const float*)d_in[12]; const float* bae1 = (const float*)d_in[13];
  const float* Wle0 = (const float*)d_in[14]; const float* ble0 = (const float*)d_in[15];
  const float* Wle1 = (const float*)d_in[16]; const float* ble1 = (const float*)d_in[17];

  char* p = (char*)d_ws;
  auto take = [&](size_t bytes) { char* q = p; p += (bytes + 255) & ~(size_t)255; return q; };
  _Float16* nodesF = (_Float16*)take((size_t)BB * NN * 64 * 2);
  _Float16* edgesF = (_Float16*)take((size_t)BB * NEDGE * 32 * 2);
  _Float16* aggN   = (_Float16*)take((size_t)BB * NEDGE * 64 * 2);
  _Float16* aggE   = (_Float16*)take((size_t)BB * NEDGE * 32 * 2);
  _Float16* prevN  = (_Float16*)take((size_t)BB * NN * 64 * 2);
  _Float16* prevE  = (_Float16*)take((size_t)BB * NEDGE * 32 * 2);
  int2*     sd     = (int2*)take((size_t)NEDGE * sizeof(int2));
  _Float16* Wan0t = (_Float16*)take(160 * 128 * 2);
  _Float16* Wan1t = (_Float16*)take(128 * 64 * 2);
  _Float16* Wln0t = (_Float16*)take(128 * 128 * 2);
  _Float16* Wln1t = (_Float16*)take(128 * 64 * 2);
  _Float16* Wae0t = (_Float16*)take(96 * 64 * 2);
  _Float16* Wae1t = (_Float16*)take(64 * 32 * 2);
  _Float16* Wle0t = (_Float16*)take(64 * 64 * 2);
  _Float16* Wle1t = (_Float16*)take(64 * 32 * 2);

  // --- setup: conversions, weight transposes, edge topology table ---
  {
    long n = (long)BB * NN * 64;
    cvt_f16<<<(unsigned)((n + 255) / 256), 256, 0, stream>>>(in_nodes, nodesF, n);
    n = (long)BB * NEDGE * 32;
    cvt_f16<<<(unsigned)((n + 255) / 256), 256, 0, stream>>>(in_edges, edgesF, n);
  }
  auto tw = [&](const float* W, _Float16* Wt, int fin, int fout) {
    transpose_w<<<(fin * fout + 255) / 256, 256, 0, stream>>>(W, Wt, fin, fout);
  };
  tw(Wan0, Wan0t, 160, 128);  tw(Wan1, Wan1t, 128, 64);
  tw(Wln0, Wln0t, 128, 128);  tw(Wln1, Wln1t, 128, 64);
  tw(Wae0, Wae0t, 96, 64);    tw(Wae1, Wae1t, 64, 32);
  tw(Wle0, Wle0t, 64, 64);    tw(Wle1, Wle1t, 64, 32);
  build_sd<<<(NEDGE + 255) / 256, 256, 0, stream>>>(sd);

  const int edge_tiles_blocks = (BB * NEDGE / 16) / 4;   // 2015
  const int node_tiles_blocks = (BB * NN / 16) / 4;      // 64

  // --- per-edge agg MLPs (WMMA) ---
  mlp2_kernel<0, 160, 128, 64, false><<<edge_tiles_blocks, 128, 0, stream>>>(
      nodesF, edgesF, nodesF, sd, Wan0t, ban0, Wan1t, ban1, aggN);
  mlp2_kernel<1, 96, 64, 32, false><<<edge_tiles_blocks, 128, 0, stream>>>(
      nodesF, edgesF, nodesF, sd, Wae0t, bae0, Wae1t, bae1, aggE);

  // --- windowed mean / exclusive prefix mean ---
  aggregate_kernel<<<dim3(NN, BB), 96, 0, stream>>>(aggN, aggE, prevN, prevE);

  // --- output MLPs (WMMA), f32 results straight into d_out ---
  float* out_nodes = (float*)d_out;
  float* out_edges = (float*)d_out + (size_t)BB * NN * 64;
  mlp2_kernel<2, 128, 128, 64, true><<<node_tiles_blocks, 128, 0, stream>>>(
      nodesF, edgesF, prevN, sd, Wln0t, bln0, Wln1t, bln1, out_nodes);
  mlp2_kernel<3, 64, 64, 32, true><<<edge_tiles_blocks, 128, 0, stream>>>(
      nodesF, edgesF, prevE, sd, Wle0t, ble0, Wle1t, ble1, out_edges);
}
